// TimeSformer_3856880632240
// MI455X (gfx1250) — matrix operand order repaired and checked by reading, not compiled, not run on hardware
//
#include <hip/hip_runtime.h>
#include <math.h>

#define B_       2
#define F_       8
#define CCH      3
#define IMG      224
#define PSZ      16
#define DIM      512
#define HEADS    8
#define DHEAD    64
#define DEPTH    12
#define NCLS     1000
#define HP_      14
#define WP_      14
#define NPATCH   196
#define PATCH_DIM 768
#define INNER    512
#define NT       1569
#define ROWS     (B_*NT)          // 3138
#define BHN      (B_*HEADS)       // 16
#define SCALE_Q  0.125f

typedef _Float16 f16;
typedef __attribute__((ext_vector_type(16))) _Float16 v16h;
typedef __attribute__((ext_vector_type(8)))  float    v8f;

union Frag { v16h v; unsigned u[8]; };
union H8   { f16 h[8]; uint4 u; };
union H4   { f16 h[4]; uint2 u; };

// ---------------------------------------------------------------------------
// Generic GEMM: C[M,N] = A[M,K] @ B[K,N] (+bias) (+residual), f32 in/out,
// f16 WMMA compute (v_wmma_f32_16x16x32_f16). K % 32 == 0, N % 64 == 0,
// M guarded only at the store (A row addresses clamped -> branch-free loop).
// Block = 256 threads = 8 waves; block tile 128x64; each wave owns a 32x32
// output tile = 4 WMMA accumulators. LDS double-buffered, 80B row stride so
// staging is ds_store_b128/b64 and fragments are ds_load_b128.
// ---------------------------------------------------------------------------
#define BM 128
#define BN 64
#define BK 32
#define LDSP 40                     // halves per row (80 B, multiple of 16)

template<bool HAS_BIAS, bool HAS_RES>
__global__ __launch_bounds__(256)
void gemm_wmma_f16(const float* __restrict__ A, const float* __restrict__ Bw,
                   const float* __restrict__ bias, const float* __restrict__ res,
                   float* __restrict__ Cm, int M, int Nn, int K)
{
  __shared__ f16 As[2][BM][LDSP];   // row-major [m][k]
  __shared__ f16 Bs[2][BN][LDSP];   // transposed [n][k]

  const int tid  = threadIdx.x;
  const int lane = tid & 31;
  const int wid  = tid >> 5;
  const int wm   = wid & 3;        // 4 wave-rows of 32
  const int wn   = wid >> 2;       // 2 wave-cols of 32
  const int rowBase = blockIdx.y * BM;
  const int colBase = blockIdx.x * BN;
  const int lm  = lane & 15;
  const int lhi = lane >> 4;

  // A staging: each thread owns one row-of-64 x 8 contiguous columns (x2 passes)
  const int ar2 = tid >> 2;        // 0..63
  const int ac2 = (tid & 3) << 3;  // 0,8,16,24
  // B staging: each thread owns 2 columns x 4 k-rows
  const int bn2 = (tid & 31) << 1; // 0..62
  const int bk4 = (tid >> 5) << 2; // 0..28

  // Clamped A row pointers (branch-free loads; OOB rows only feed OOB C rows)
  const float* aptr[2];
#pragma unroll
  for (int g = 0; g < 2; ++g) {
    int grow = rowBase + ar2 + (g << 6);
    if (grow > M - 1) grow = M - 1;
    aptr[g] = A + (size_t)grow * K + ac2;
  }
  const float* bptr[4];
#pragma unroll
  for (int kk = 0; kk < 4; ++kk)
    bptr[kk] = Bw + (size_t)(bk4 + kk) * Nn + colBase + bn2;

  v8f acc00 = {0,0,0,0,0,0,0,0};
  v8f acc01 = {0,0,0,0,0,0,0,0};
  v8f acc10 = {0,0,0,0,0,0,0,0};
  v8f acc11 = {0,0,0,0,0,0,0,0};

  const int NC = K / BK;
  float4 a0[2], a1[2];
  float2 bq[4];

  // --- preload chunk 0 and stage into LDS buffer 0 ---
#pragma unroll
  for (int g = 0; g < 2; ++g) {
    a0[g] = *reinterpret_cast<const float4*>(aptr[g]);
    a1[g] = *reinterpret_cast<const float4*>(aptr[g] + 4);
  }
#pragma unroll
  for (int kk = 0; kk < 4; ++kk)
    bq[kk] = *reinterpret_cast<const float2*>(bptr[kk]);

#pragma unroll
  for (int g = 0; g < 2; ++g) {
    H8 pk;
    pk.h[0] = (f16)a0[g].x; pk.h[1] = (f16)a0[g].y;
    pk.h[2] = (f16)a0[g].z; pk.h[3] = (f16)a0[g].w;
    pk.h[4] = (f16)a1[g].x; pk.h[5] = (f16)a1[g].y;
    pk.h[6] = (f16)a1[g].z; pk.h[7] = (f16)a1[g].w;
    *reinterpret_cast<uint4*>(&As[0][ar2 + (g << 6)][ac2]) = pk.u;
  }
  {
    H4 px, py;
#pragma unroll
    for (int kk = 0; kk < 4; ++kk) { px.h[kk] = (f16)bq[kk].x; py.h[kk] = (f16)bq[kk].y; }
    *reinterpret_cast<uint2*>(&Bs[0][bn2][bk4])     = px.u;
    *reinterpret_cast<uint2*>(&Bs[0][bn2 + 1][bk4]) = py.u;
  }
  __syncthreads();

  for (int c = 0; c < NC; ++c) {
    const int cur = c & 1;
    // --- prefetch chunk c+1 into registers ---
    if (c + 1 < NC) {
      int k0 = (c + 1) * BK;
#pragma unroll
      for (int g = 0; g < 2; ++g) {
        a0[g] = *reinterpret_cast<const float4*>(aptr[g] + k0);
        a1[g] = *reinterpret_cast<const float4*>(aptr[g] + k0 + 4);
      }
#pragma unroll
      for (int kk = 0; kk < 4; ++kk)
        bq[kk] = *reinterpret_cast<const float2*>(bptr[kk] + (size_t)k0 * Nn);
    }

    // --- fragments from LDS (contiguous 16B uint groups), ISA 7.12.2 layouts ---
    const unsigned* ap0 = reinterpret_cast<const unsigned*>(&As[cur][(wm << 5) + lm][0]);
    const unsigned* ap1 = reinterpret_cast<const unsigned*>(&As[cur][(wm << 5) + 16 + lm][0]);
    const unsigned* bp0 = reinterpret_cast<const unsigned*>(&Bs[cur][(wn << 5) + lm][0]);
    const unsigned* bp1 = reinterpret_cast<const unsigned*>(&Bs[cur][(wn << 5) + 16 + lm][0]);
    Frag af0, af1, bf0, bf1;
#pragma unroll
    for (int r = 0; r < 4; ++r) {
      af0.u[r]     = ap0[(lhi << 2) + r];
      af0.u[4 + r] = ap0[8 + (lhi << 2) + r];
      af1.u[r]     = ap1[(lhi << 2) + r];
      af1.u[4 + r] = ap1[8 + (lhi << 2) + r];
    }
#pragma unroll
    for (int r = 0; r < 8; ++r) {
      bf0.u[r] = bp0[((r >> 2) << 3) + (lhi << 2) + (r & 3)];
      bf1.u[r] = bp1[((r >> 2) << 3) + (lhi << 2) + (r & 3)];
    }

    acc00 = __builtin_amdgcn_wmma_f32_16x16x32_f16(false, af0.v, false, bf0.v, (short)0, acc00, false, false);
    acc01 = __builtin_amdgcn_wmma_f32_16x16x32_f16(false, af0.v, false, bf1.v, (short)0, acc01, false, false);
    acc10 = __builtin_amdgcn_wmma_f32_16x16x32_f16(false, af1.v, false, bf0.v, (short)0, acc10, false, false);
    acc11 = __builtin_amdgcn_wmma_f32_16x16x32_f16(false, af1.v, false, bf1.v, (short)0, acc11, false, false);

    // --- stage chunk c+1 into the other LDS buffer ---
    if (c + 1 < NC) {
      const int nxt = cur ^ 1;
#pragma unroll
      for (int g = 0; g < 2; ++g) {
        H8 pk;
        pk.h[0] = (f16)a0[g].x; pk.h[1] = (f16)a0[g].y;
        pk.h[2] = (f16)a0[g].z; pk.h[3] = (f16)a0[g].w;
        pk.h[4] = (f16)a1[g].x; pk.h[5] = (f16)a1[g].y;
        pk.h[6] = (f16)a1[g].z; pk.h[7] = (f16)a1[g].w;
        *reinterpret_cast<uint4*>(&As[nxt][ar2 + (g << 6)][ac2]) = pk.u;
      }
      H4 px, py;
#pragma unroll
      for (int kk = 0; kk < 4; ++kk) { px.h[kk] = (f16)bq[kk].x; py.h[kk] = (f16)bq[kk].y; }
      *reinterpret_cast<uint2*>(&Bs[nxt][bn2][bk4])     = px.u;
      *reinterpret_cast<uint2*>(&Bs[nxt][bn2 + 1][bk4]) = py.u;
      __syncthreads();
    }
  }

  // --- epilogue: C/D layout: VGPR r -> row r + 8*lhi, col = lane&15 ---
  const v8f* accs[4] = { &acc00, &acc01, &acc10, &acc11 };
#pragma unroll
  for (int i = 0; i < 2; ++i) {
#pragma unroll
    for (int j = 0; j < 2; ++j) {
      const v8f& a = *accs[i * 2 + j];
      int col = colBase + (wn << 5) + (j << 4) + lm;
      float bcol = HAS_BIAS ? bias[col] : 0.0f;
#pragma unroll
      for (int r = 0; r < 8; ++r) {
        int row = rowBase + (wm << 5) + (i << 4) + r + (lhi << 3);
        if (row < M) {
          float v = a[r] + bcol;
          if (HAS_RES) v += res[(size_t)row * Nn + col];
          Cm[(size_t)row * Nn + col] = v;
        }
      }
    }
  }
}

static inline void launch_gemm(const float* A, const float* Bw, const float* bias,
                               const float* res, float* Cm, int M, int Nn, int K,
                               hipStream_t s)
{
  dim3 grid(Nn / BN, (M + BM - 1) / BM);
  if (bias && res)
    gemm_wmma_f16<true, true><<<grid, 256, 0, s>>>(A, Bw, bias, res, Cm, M, Nn, K);
  else if (bias)
    gemm_wmma_f16<true, false><<<grid, 256, 0, s>>>(A, Bw, bias, res, Cm, M, Nn, K);
  else
    gemm_wmma_f16<false, false><<<grid, 256, 0, s>>>(A, Bw, bias, res, Cm, M, Nn, K);
}

// ---------------------------------------------------------------------------
// Rotary tables: frame (F x 64) and image (196 x 64), repeat-2 layout baked in.
// ---------------------------------------------------------------------------
__global__ void rot_init_k(float* fs, float* fc, float* is_, float* ic_)
{
  int tid = threadIdx.x;
  const float PI = 3.14159265358979323846f;
  for (int i = tid; i < F_ * 32; i += 256) {
    int f = i >> 5, k2 = i & 31;
    float inv = powf(10000.0f, -(float)k2 / 32.0f);
    float a = (float)f * inv;
    float s = sinf(a), c = cosf(a);
    fs[f * 64 + 2 * k2] = s; fs[f * 64 + 2 * k2 + 1] = s;
    fc[f * 64 + 2 * k2] = c; fc[f * 64 + 2 * k2 + 1] = c;
  }
  for (int i = tid; i < NPATCH * 16; i += 256) {
    int n = i >> 4, j = i & 15;
    int h = n / WP_, w = n % WP_;
    float sc2 = powf(5.0f, (float)j / 15.0f);
    float xa = (-1.0f + 2.0f * (float)h / 13.0f) * sc2 * PI;
    float ya = (-1.0f + 2.0f * (float)w / 13.0f) * sc2 * PI;
    float sx = sinf(xa), sy = sinf(ya), cx = cosf(xa), cy = cosf(ya);
    is_[n * 64 + 2 * j] = sx;      is_[n * 64 + 2 * j + 1] = sx;
    is_[n * 64 + 32 + 2 * j] = sy; is_[n * 64 + 32 + 2 * j + 1] = sy;
    ic_[n * 64 + 2 * j] = cx;      ic_[n * 64 + 2 * j + 1] = cx;
    ic_[n * 64 + 32 + 2 * j] = cy; ic_[n * 64 + 32 + 2 * j + 1] = cy;
  }
}

// video (B,F,C,224,224) -> patches (B*F*196, 768) with inner layout (py*16+px)*3+c
__global__ void patchify_k(const float* __restrict__ video, float* __restrict__ patches)
{
  int idx = blockIdx.x * 256 + threadIdx.x;
  if (idx >= B_ * F_ * NPATCH * PATCH_DIM) return;
  int pc   = idx % PATCH_DIM;
  int prow = idx / PATCH_DIM;
  int n    = prow % NPATCH;
  int bf   = prow / NPATCH;          // b*F + f
  int h = n / WP_, w = n % WP_;
  int c   = pc % CCH;
  int pix = pc / CCH;                // py*16+px
  int py = pix >> 4, px = pix & 15;
  size_t vidx = (((size_t)bf * CCH + c) * IMG + (h * PSZ + py)) * IMG + (w * PSZ + px);
  patches[idx] = video[vidx];
}

// x[b, 0] = cls ; x[b, 1+t] = emb[b*F*N + t]
__global__ void build_x_k(const float* __restrict__ emb, const float* __restrict__ cls,
                          float* __restrict__ x)
{
  int idx = blockIdx.x * 256 + threadIdx.x;
  if (idx >= ROWS * DIM) return;
  int col = idx % DIM;
  int row = idx / DIM;
  int b = row / NT, t = row % NT;
  x[idx] = (t == 0) ? cls[col]
                    : emb[((size_t)(b * F_ * NPATCH + (t - 1))) * DIM + col];
}

// LayerNorm over DIM=512; input row index = blockIdx.x * rowStride
__global__ __launch_bounds__(256)
void layernorm_k(const float* __restrict__ in, const float* __restrict__ g,
                 const float* __restrict__ b, float* __restrict__ out, int rowStride)
{
  int row = blockIdx.x;
  int tid = threadIdx.x;
  const float* p = in + (size_t)row * rowStride * DIM;
  float* o = out + (size_t)row * DIM;
  __shared__ float red[256];
  float v0 = p[tid], v1 = p[tid + 256];
  red[tid] = v0 + v1;
  __syncthreads();
  for (int s = 128; s > 0; s >>= 1) { if (tid < s) red[tid] += red[tid + s]; __syncthreads(); }
  float mean = red[0] * (1.0f / 512.0f);
  __syncthreads();
  float d0 = v0 - mean, d1 = v1 - mean;
  red[tid] = d0 * d0 + d1 * d1;
  __syncthreads();
  for (int s = 128; s > 0; s >>= 1) { if (tid < s) red[tid] += red[tid + s]; __syncthreads(); }
  float rstd = rsqrtf(red[0] * (1.0f / 512.0f) + 1e-5f);
  o[tid]       = d0 * rstd * g[tid] + b[tid];
  o[tid + 256] = d1 * rstd * g[tid + 256] + b[tid + 256];
}

// qkv (ROWS,1536) -> per-head qs (rot+scale), ks (rot), ksr (raw), vs.
// mode 0 = time (rot idx f), mode 1 = space (rot idx n). cls token unrotated.
__global__ void qkv_prep_k(const float* __restrict__ qkv,
                           const float* __restrict__ fs, const float* __restrict__ fc,
                           const float* __restrict__ is_, const float* __restrict__ ic_,
                           float* qs, float* ks, float* ksr, float* vs, int mode)
{
  int idx = blockIdx.x * 256 + threadIdx.x;
  if (idx >= BHN * NT * 32) return;
  int pr   = idx & 31;
  int rest = idx >> 5;
  int t  = rest % NT;
  int bh = rest / NT;
  int b = bh / HEADS, h = bh % HEADS;
  int d0 = pr * 2;
  const float* src = qkv + (size_t)(b * NT + t) * 1536;
  float qe = src[h * 64 + d0] * SCALE_Q, qo = src[h * 64 + d0 + 1] * SCALE_Q;
  float ke = src[512 + h * 64 + d0],    ko = src[512 + h * 64 + d0 + 1];
  float ve = src[1024 + h * 64 + d0],   vo = src[1024 + h * 64 + d0 + 1];
  size_t o = ((size_t)bh * NT + t) * 64 + d0;
  ksr[o] = ke; ksr[o + 1] = ko;
  vs[o]  = ve; vs[o + 1]  = vo;
  if (t > 0) {
    int tok = t - 1;
    int fi = tok / NPATCH, n = tok % NPATCH;
    int ri = (mode == 0) ? fi : n;
    const float* sn = (mode == 0 ? fs : is_) + ri * 64;
    const float* cn = (mode == 0 ? fc : ic_) + ri * 64;
    float c = cn[d0], s = sn[d0];
    float nqe = qe * c - qo * s, nqo = qo * c + qe * s;
    float nke = ke * c - ko * s, nko = ko * c + ke * s;
    qe = nqe; qo = nqo; ke = nke; ko = nko;
  }
  qs[o] = qe; qs[o + 1] = qo;
  ks[o] = ke; ks[o + 1] = ko;
}

// cls query attends over full (unrotated) k / v sequence. One block per (b,h).
__global__ __launch_bounds__(256)
void cls_attn_k(const float* __restrict__ qs, const float* __restrict__ ksr,
                const float* __restrict__ vs, float* __restrict__ attno)
{
  int bh = blockIdx.x;
  int tid = threadIdx.x;
  __shared__ float sc[NT];
  __shared__ float red[256];
  __shared__ float qv[64];
  const size_t base = (size_t)bh * NT * 64;
  if (tid < 64) qv[tid] = qs[base + tid];
  __syncthreads();
  for (int j = tid; j < NT; j += 256) {
    const float* kp = ksr + base + (size_t)j * 64;
    float s = 0.f;
    for (int d = 0; d < 64; ++d) s += qv[d] * kp[d];
    sc[j] = s;
  }
  __syncthreads();
  float m = -3.4e38f;
  for (int j = tid; j < NT; j += 256) m = fmaxf(m, sc[j]);
  red[tid] = m; __syncthreads();
  for (int s2 = 128; s2 > 0; s2 >>= 1) { if (tid < s2) red[tid] = fmaxf(red[tid], red[tid + s2]); __syncthreads(); }
  float mx = red[0]; __syncthreads();
  float part = 0.f;
  for (int j = tid; j < NT; j += 256) { float e = expf(sc[j] - mx); sc[j] = e; part += e; }
  red[tid] = part; __syncthreads();
  for (int s2 = 128; s2 > 0; s2 >>= 1) { if (tid < s2) red[tid] += red[tid + s2]; __syncthreads(); }
  float dn = red[0]; __syncthreads();
  int d = tid & 63, grp = tid >> 6;
  float accv = 0.f;
  for (int j = grp; j < NT; j += 4) accv += sc[j] * vs[base + (size_t)j * 64 + d];
  red[tid] = accv; __syncthreads();
  if (tid < 64) {
    float o = (red[tid] + red[tid + 64] + red[tid + 128] + red[tid + 192]) / dn;
    int b = bh / HEADS, h = bh % HEADS;
    attno[(size_t)(b * NT) * INNER + h * 64 + tid] = o;
  }
}

// Time attention: block per (bh, n): 8 queries over 9 keys (cls + 8 frames).
__global__ __launch_bounds__(128)
void time_attn_k(const float* __restrict__ qs, const float* __restrict__ ks,
                 const float* __restrict__ vs, float* __restrict__ attno)
{
  int blk = blockIdx.x;
  int bh = blk / NPATCH, n = blk % NPATCH;
  int tid = threadIdx.x;
  __shared__ float q8[8][64], kk[9][64], vv[9][64], sc[8][9], pr2[8][9];
  const size_t base = (size_t)bh * NT * 64;
  for (int i = tid; i < 8 * 64; i += 128) {
    int f = i >> 6, d = i & 63;
    q8[f][d] = qs[base + (size_t)(1 + f * NPATCH + n) * 64 + d];
  }
  for (int i = tid; i < 9 * 64; i += 128) {
    int j = i >> 6, d = i & 63;
    int t = (j == 0) ? 0 : (1 + (j - 1) * NPATCH + n);
    kk[j][d] = ks[base + (size_t)t * 64 + d];
    vv[j][d] = vs[base + (size_t)t * 64 + d];
  }
  __syncthreads();
  if (tid < 72) {
    int f = tid / 9, j = tid % 9;
    float s = 0.f;
    for (int d = 0; d < 64; ++d) s += q8[f][d] * kk[j][d];
    sc[f][j] = s;
  }
  __syncthreads();
  if (tid < 8) {
    float mx = -3.4e38f;
    for (int j = 0; j < 9; ++j) mx = fmaxf(mx, sc[tid][j]);
    float sum = 0.f;
    for (int j = 0; j < 9; ++j) { float e = expf(sc[tid][j] - mx); pr2[tid][j] = e; sum += e; }
    float inv = 1.f / sum;
    for (int j = 0; j < 9; ++j) pr2[tid][j] *= inv;
  }
  __syncthreads();
  if (tid < 64) {
    int b = bh / HEADS, h = bh % HEADS;
    for (int f = 0; f < 8; ++f) {
      float o = 0.f;
      for (int j = 0; j < 9; ++j) o += pr2[f][j] * vv[j][tid];
      attno[(size_t)(b * NT + 1 + f * NPATCH + n) * INNER + h * 64 + tid] = o;
    }
  }
}

// Space attention: block per (bh, f): 196 queries over 197 keys (cls + 196).
__global__ __launch_bounds__(256)
void space_attn_k(const float* __restrict__ qs, const float* __restrict__ ks,
                  const float* __restrict__ vs, float* __restrict__ attno)
{
  int blk = blockIdx.x;
  int bh = blk / F_, f = blk % F_;
  int b = bh / HEADS, h = bh % HEADS;
  int tid = threadIdx.x;
  __shared__ float qv[64], sc[224], red[256];
  const size_t base = (size_t)bh * NT * 64;
  for (int q = 0; q < NPATCH; ++q) {
    int qt = 1 + f * NPATCH + q;
    if (tid < 64) qv[tid] = qs[base + (size_t)qt * 64 + tid];
    __syncthreads();
    if (tid < 197) {
      int t = (tid == 0) ? 0 : (1 + f * NPATCH + tid - 1);
      const float* kp = ks + base + (size_t)t * 64;
      float s = 0.f;
      for (int d = 0; d < 64; ++d) s += qv[d] * kp[d];
      sc[tid] = s;
    }
    __syncthreads();
    red[tid] = (tid < 197) ? sc[tid] : -3.4e38f;
    __syncthreads();
    for (int s2 = 128; s2 > 0; s2 >>= 1) { if (tid < s2) red[tid] = fmaxf(red[tid], red[tid + s2]); __syncthreads(); }
    float mx = red[0]; __syncthreads();
    float e = 0.f;
    if (tid < 197) e = expf(sc[tid] - mx);
    red[tid] = e; __syncthreads();
    for (int s2 = 128; s2 > 0; s2 >>= 1) { if (tid < s2) red[tid] += red[tid + s2]; __syncthreads(); }
    float dn = red[0]; __syncthreads();
    if (tid < 197) sc[tid] = e / dn;
    __syncthreads();
    int d = tid & 63, grp = tid >> 6;
    float accv = 0.f;
    for (int j = grp; j < 197; j += 4) {
      int t = (j == 0) ? 0 : (1 + f * NPATCH + j - 1);
      accv += sc[j] * vs[base + (size_t)t * 64 + d];
    }
    red[tid] = accv; __syncthreads();
    if (tid < 64) {
      float o = red[tid] + red[tid + 64] + red[tid + 128] + red[tid + 192];
      attno[(size_t)(b * NT + qt) * INNER + h * 64 + tid] = o;
    }
    __syncthreads();
  }
}

// GeGLU: u = a * gelu_exact(g), h = [a | g] along cols
__global__ void geglu_k(const float* __restrict__ hb, float* __restrict__ ub)
{
  int idx = blockIdx.x * 256 + threadIdx.x;
  if (idx >= ROWS * 2048) return;
  int col = idx % 2048;
  int row = idx / 2048;
  float a = hb[(size_t)row * 4096 + col];
  float g = hb[(size_t)row * 4096 + 2048 + col];
  float ge = 0.5f * g * (1.0f + erff(g * 0.7071067811865476f));
  ub[idx] = a * ge;
}

// head: out[b,c] = ln[b] . head_w[:,c] + head_b[c]
__global__ void head_k(const float* __restrict__ ln, const float* __restrict__ hw,
                       const float* __restrict__ hb, float* __restrict__ out)
{
  int c = blockIdx.x * 256 + threadIdx.x;
  int b = blockIdx.y;
  if (c >= NCLS) return;
  float s = hb[c];
  const float* r = ln + b * DIM;
  for (int d = 0; d < DIM; ++d) s += r[d] * hw[(size_t)d * NCLS + c];
  out[b * NCLS + c] = s;
}

// ---------------------------------------------------------------------------
extern "C" void kernel_launch(void* const* d_in, const int* in_sizes, int n_in,
                              void* d_out, int out_size, void* d_ws, size_t ws_size,
                              hipStream_t stream)
{
  (void)in_sizes; (void)n_in; (void)out_size; (void)ws_size;
  const float* video   = (const float*)d_in[0];
  const float* patch_w = (const float*)d_in[1];
  const float* patch_b = (const float*)d_in[2];
  const float* cls_tok = (const float*)d_in[3];
  const float* ln_t_g  = (const float*)d_in[4];
  const float* ln_t_b  = (const float*)d_in[5];
  const float* qkv_t   = (const float*)d_in[6];
  const float* out_t_w = (const float*)d_in[7];
  const float* out_t_b = (const float*)d_in[8];
  const float* ln_s_g  = (const float*)d_in[9];
  const float* ln_s_b  = (const float*)d_in[10];
  const float* qkv_s   = (const float*)d_in[11];
  const float* out_s_w = (const float*)d_in[12];
  const float* out_s_b = (const float*)d_in[13];
  const float* ln_f_g  = (const float*)d_in[14];
  const float* ln_f_b  = (const float*)d_in[15];
  const float* ff_w1   = (const float*)d_in[16];
  const float* ff_b1   = (const float*)d_in[17];
  const float* ff_w2   = (const float*)d_in[18];
  const float* ff_b2   = (const float*)d_in[19];
  const float* ln_o_g  = (const float*)d_in[20];
  const float* ln_o_b  = (const float*)d_in[21];
  const float* head_w  = (const float*)d_in[22];
  const float* head_b  = (const float*)d_in[23];
  float* out = (float*)d_out;
  float* ws  = (float*)d_ws;

  float* patches = ws + 0;          // 2,408,448
  float* emb     = ws + 2408448;    // 1,605,632
  float* x       = ws + 4014080;    // 1,606,656
  float* xn      = ws + 5620736;    // 1,606,656
  float* qkvb    = ws + 7227392;    // 4,819,968
  float* qs      = ws + 12047360;   // 1,606,656
  float* ks      = ws + 13654016;
  float* ksr     = ws + 15260672;
  float* vs      = ws + 16867328;
  float* attno   = ws + 18473984;
  float* hbuf    = ws + 20080640;   // 12,853,248
  float* ubuf    = ws + 32933888;   // 6,426,624
  float* fsin    = ws + 39360512;
  float* fcos    = fsin + 512;
  float* isin    = fcos + 512;
  float* icos    = isin + 12544;
  float* clsn    = icos + 12544;

  rot_init_k<<<1, 256, 0, stream>>>(fsin, fcos, isin, icos);
  patchify_k<<<9408, 256, 0, stream>>>(video, patches);
  launch_gemm(patches, patch_w, patch_b, nullptr, emb, 3136, 512, 768, stream);
  build_x_k<<<6276, 256, 0, stream>>>(emb, cls_tok, x);

  for (int i = 0; i < DEPTH; ++i) {
    // ---- time attention block ----
    layernorm_k<<<ROWS, 256, 0, stream>>>(x, ln_t_g + (size_t)i * DIM, ln_t_b + (size_t)i * DIM, xn, 1);
    launch_gemm(xn, qkv_t + (size_t)i * DIM * 1536, nullptr, nullptr, qkvb, ROWS, 1536, DIM, stream);
    qkv_prep_k<<<3138, 256, 0, stream>>>(qkvb, fsin, fcos, isin, icos, qs, ks, ksr, vs, 0);
    cls_attn_k<<<BHN, 256, 0, stream>>>(qs, ksr, vs, attno);
    time_attn_k<<<BHN * NPATCH, 128, 0, stream>>>(qs, ks, vs, attno);
    launch_gemm(attno, out_t_w + (size_t)i * DIM * DIM, out_t_b + (size_t)i * DIM, x, x, ROWS, DIM, DIM, stream);
    // ---- space attention block ----
    layernorm_k<<<ROWS, 256, 0, stream>>>(x, ln_s_g + (size_t)i * DIM, ln_s_b + (size_t)i * DIM, xn, 1);
    launch_gemm(xn, qkv_s + (size_t)i * DIM * 1536, nullptr, nullptr, qkvb, ROWS, 1536, DIM, stream);
    qkv_prep_k<<<3138, 256, 0, stream>>>(qkvb, fsin, fcos, isin, icos, qs, ks, ksr, vs, 1);
    cls_attn_k<<<BHN, 256, 0, stream>>>(qs, ksr, vs, attno);
    space_attn_k<<<BHN * F_, 256, 0, stream>>>(qs, ks, vs, attno);
    launch_gemm(attno, out_s_w + (size_t)i * DIM * DIM, out_s_b + (size_t)i * DIM, x, x, ROWS, DIM, DIM, stream);
    // ---- GeGLU FF block ----
    layernorm_k<<<ROWS, 256, 0, stream>>>(x, ln_f_g + (size_t)i * DIM, ln_f_b + (size_t)i * DIM, xn, 1);
    launch_gemm(xn, ff_w1 + (size_t)i * DIM * 4096, ff_b1 + (size_t)i * 4096, nullptr, hbuf, ROWS, 4096, DIM, stream);
    geglu_k<<<25104, 256, 0, stream>>>(hbuf, ubuf);
    launch_gemm(ubuf, ff_w2 + (size_t)i * 2048 * DIM, ff_b2 + (size_t)i * DIM, x, x, ROWS, DIM, 2048, stream);
  }

  layernorm_k<<<2, 256, 0, stream>>>(x, ln_o_g, ln_o_b, clsn, NT);
  head_k<<<dim3(4, 2), 256, 0, stream>>>(clsn, head_w, head_b, out);
}
